// MyModel_19851338842195
// MI455X (gfx1250) — compile-verified, matrix operand order checked by reading, not run on hardware
//
#include <hip/hip_runtime.h>
#include <hip/hip_bf16.h>
#include <math.h>

// ---------------------------------------------------------------------------
// MI455X (gfx1250, wave32) implementation of the multi-modal molecule model.
// Dense layers: f16 WMMA (v_wmma_f32_16x16x32_f16) with f32 accumulation.
// Operands are pre-packed into WMMA fragment layout (f16); each wave computes
// a 16x32 output slab (one A fragment reused across two B fragments / two
// accumulators), so the MAC loop is 3 contiguous v16h loads + 2 wmma with no
// branches. Attention / GAT use online softmax; LN/BN use wave32 shuffles.
// ---------------------------------------------------------------------------

typedef __attribute__((ext_vector_type(16))) _Float16 v16h;
typedef __attribute__((ext_vector_type(8)))  float    v8f;

__device__ __forceinline__ float wave_sum(float v) {
#pragma unroll
  for (int off = 16; off > 0; off >>= 1) v += __shfl_xor(v, off, 32);
  return v;
}

// ---------------------------------------------------------------------------
// Operand packing into WMMA fragment layout (f16, zero-padded past K).
// A-frag: lane l, elem e -> A[tm*16 + (l&15)][kt*32 + (l>>4)*16 + e]
// B-frag: lane l, elem e -> B[kt*32 + (l>>4)*16 + e][tn*16 + (l&15)]
// Packed index: ((kt*tiles + t)*32 + lane)*16 + e   (32B-aligned per lane)
// ---------------------------------------------------------------------------
__global__ void pack_a_kernel(const float* __restrict__ A, _Float16* __restrict__ Ap,
                              int M, int K) {
  const int tilesM = M >> 4;
  const int Kt = (K + 31) >> 5;
  int idx = blockIdx.x * blockDim.x + threadIdx.x;  // (kt*tilesM + tm)*32 + lane
  if (idx >= Kt * tilesM * 32) return;
  const int lane = idx & 31;
  const int tm = (idx >> 5) % tilesM;
  const int kt = (idx >> 5) / tilesM;
  const float* __restrict__ src = A + (size_t)(tm * 16 + (lane & 15)) * K;
  const int kbase = kt * 32 + (lane >> 4) * 16;
  v16h v;
  if (kbase + 16 <= K) {
#pragma unroll
    for (int e = 0; e < 16; ++e) v[e] = (_Float16)src[kbase + e];
  } else {
#pragma unroll
    for (int e = 0; e < 16; ++e) {
      int k = kbase + e;
      v[e] = (k < K) ? (_Float16)src[k] : (_Float16)0.0f;
    }
  }
  *((v16h*)(Ap + (size_t)idx * 16)) = v;
}

__global__ void pack_b_kernel(const float* __restrict__ B, _Float16* __restrict__ Bp,
                              int K, int N) {
  const int tilesN = N >> 4;  // N is a multiple of 32 at every call site
  const int Kt = (K + 31) >> 5;
  int idx = blockIdx.x * blockDim.x + threadIdx.x;  // (kt*tilesN + tn)*32 + lane
  if (idx >= Kt * tilesN * 32) return;
  const int lane = idx & 31;
  const int tn = (idx >> 5) % tilesN;
  const int kt = (idx >> 5) / tilesN;
  const int col = tn * 16 + (lane & 15);
  const int kbase = kt * 32 + (lane >> 4) * 16;
  v16h v;
  if (kbase + 16 <= K) {
#pragma unroll
    for (int e = 0; e < 16; ++e) v[e] = (_Float16)B[(size_t)(kbase + e) * N + col];
  } else {
#pragma unroll
    for (int e = 0; e < 16; ++e) {
      int k = kbase + e;
      v[e] = (k < K) ? (_Float16)B[(size_t)k * N + col] : (_Float16)0.0f;
    }
  }
  *((v16h*)(Bp + (size_t)idx * 16)) = v;
}

// ---------------------------------------------------------------------------
// C[M,N] = act(Ap @ Bp + bias); one wave per 16x32 slab (2 accumulators,
// A fragment reused). N must be a multiple of 32 (true at every call site).
// C/D layout: lane l, vgpr j -> C[tm*16 + (l>>4)*8 + j][tn*16 + (l&15)]
// ---------------------------------------------------------------------------
template <int ACT>  // 0 = none, 1 = ReLU
__global__ void gemm_wmma_kernel(const _Float16* __restrict__ Ap,
                                 const _Float16* __restrict__ Bp,
                                 const float* __restrict__ bias,
                                 float* __restrict__ C,
                                 int M, int N, int Kt) {
  const int lane  = threadIdx.x & 31;
  const int gwave = (int)((blockIdx.x * blockDim.x + threadIdx.x) >> 5);
  const int tilesN  = N >> 4;
  const int tilesN2 = N >> 5;
  const int tilesM  = M >> 4;
  if (gwave >= tilesM * tilesN2) return;  // slab-granular: EXEC all-ones below
  const int tm  = gwave / tilesN2;
  const int tn2 = gwave % tilesN2;

  const v16h* __restrict__ ap = (const v16h*)Ap + ((size_t)tm * 32 + lane);
  const v16h* __restrict__ bp = (const v16h*)Bp + ((size_t)tn2 * 64 + lane);
  const size_t aStep = (size_t)tilesM * 32;
  const size_t bStep = (size_t)tilesN * 32;

  v8f acc0 = {}, acc1 = {};
  for (int kt = 0; kt < Kt; ++kt) {
    v16h a  = ap[0];
    v16h b0 = bp[0];
    v16h b1 = bp[32];
    // Speculative prefetch of next K-tile fragments (dropped if OOB).
    __builtin_prefetch(ap + aStep, 0, 0);
    __builtin_prefetch(bp + bStep, 0, 0);
    __builtin_prefetch(bp + bStep + 32, 0, 0);
    ap += aStep;
    bp += bStep;
    acc0 = __builtin_amdgcn_wmma_f32_16x16x32_f16(
        false, a, false, b0, (short)0, acc0, false, false);
    acc1 = __builtin_amdgcn_wmma_f32_16x16x32_f16(
        false, a, false, b1, (short)0, acc1, false, false);
  }

  const int cn0 = tn2 * 32 + (lane & 15);
  const int cn1 = cn0 + 16;
  const float bi0 = bias ? bias[cn0] : 0.0f;
  const float bi1 = bias ? bias[cn1] : 0.0f;
#pragma unroll
  for (int j = 0; j < 8; ++j) {
    int cm = tm * 16 + (lane >> 4) * 8 + j;
    float v0 = acc0[j] + bi0;
    float v1 = acc1[j] + bi1;
    if (ACT == 1) { v0 = fmaxf(v0, 0.0f); v1 = fmaxf(v1, 0.0f); }
    C[(size_t)cm * N + cn0] = v0;
    C[(size_t)cm * N + cn1] = v1;
  }
}

// --------------------------- elementwise kernels ---------------------------

__global__ void embed_pe_kernel(const int* __restrict__ tok,
                                const float* __restrict__ emb,
                                float* __restrict__ x, int B, int L, int E) {
  int idx = blockIdx.x * blockDim.x + threadIdx.x;
  if (idx >= B * L * E) return;
  int e = idx % E;
  int l = (idx / E) % L;
  int t = tok[idx / E];
  int i2 = (e >> 1) << 1;
  float div = __expf(-(float)i2 * (logf(10000.0f) / (float)E));
  float ang = (float)l * div;
  float pe = (e & 1) ? __cosf(ang) : __sinf(ang);
  x[idx] = emb[t * E + e] + pe;
}

// MHA over qkv rows (B*L, 384): per-thread (b,h,q), online softmax, d=16.
__global__ void attention_kernel(const float* __restrict__ qkv,
                                 float* __restrict__ out, int B, int L) {
  const int H = 8, D = 16, E = 128, C3 = 384;
  int idx = blockIdx.x * blockDim.x + threadIdx.x;
  if (idx >= B * H * L) return;
  int q = idx % L;
  int h = (idx / L) % H;
  int b = idx / (L * H);
  const float scale = 0.25f;  // 1/sqrt(16)
  const float* qv = qkv + ((size_t)(b * L + q)) * C3 + h * D;
  float qr[16], o[16];
#pragma unroll
  for (int t = 0; t < 16; ++t) { qr[t] = qv[t]; o[t] = 0.0f; }
  float m = -3.4e38f, s = 0.0f;
  for (int j = 0; j < L; ++j) {
    const float* kr = qkv + ((size_t)(b * L + j)) * C3 + E + h * D;
    const float* vr = kr + E;
    float e = 0.0f;
#pragma unroll
    for (int t = 0; t < 16; ++t) e += qr[t] * kr[t];
    e *= scale;
    if (e > m) {
      float f = __expf(m - e);
      s *= f;
#pragma unroll
      for (int t = 0; t < 16; ++t) o[t] *= f;
      m = e;
    }
    float w = __expf(e - m);
    s += w;
#pragma unroll
    for (int t = 0; t < 16; ++t) o[t] += w * vr[t];
  }
  float inv = 1.0f / s;
  float* op = out + ((size_t)(b * L + q)) * E + h * D;
#pragma unroll
  for (int t = 0; t < 16; ++t) op[t] = o[t] * inv;
}

// x[row] = LN(x[row] + y[row]); one wave (32 lanes) per row of 128.
__global__ void add_ln_kernel(float* __restrict__ x, const float* __restrict__ y,
                              const float* __restrict__ g, const float* __restrict__ b,
                              int rows) {
  const int E = 128;
  int wid = (int)((blockIdx.x * blockDim.x + threadIdx.x) >> 5);
  int lane = threadIdx.x & 31;
  if (wid >= rows) return;
  float* xr = x + (size_t)wid * E;
  const float* yr = y + (size_t)wid * E;
  float v[4], sum = 0.0f;
#pragma unroll
  for (int t = 0; t < 4; ++t) { v[t] = xr[lane * 4 + t] + yr[lane * 4 + t]; sum += v[t]; }
  float mean = wave_sum(sum) * (1.0f / E);
  float var = 0.0f;
#pragma unroll
  for (int t = 0; t < 4; ++t) { float d = v[t] - mean; var += d * d; }
  var = wave_sum(var) * (1.0f / E);
  float rstd = rsqrtf(var + 1e-5f);
#pragma unroll
  for (int t = 0; t < 4; ++t) {
    int e = lane * 4 + t;
    xr[e] = (v[t] - mean) * rstd * g[e] + b[e];
  }
}

__global__ void mean_seq_kernel(const float* __restrict__ x, float* __restrict__ out,
                                int B, int L, int E) {
  int idx = blockIdx.x * blockDim.x + threadIdx.x;
  if (idx >= B * E) return;
  int b = idx / E, e = idx % E;
  float s = 0.0f;
  for (int l = 0; l < L; ++l) s += x[((size_t)(b * L + l)) * E + e];
  out[idx] = s * (1.0f / (float)L);
}

// esrc = h @ a[:32], edst = h @ a[32:64]  (per node-row, F=32)
__global__ void gat_edge_kernel(const float* __restrict__ h, const float* __restrict__ a,
                                float* __restrict__ esrc, float* __restrict__ edst, int rows) {
  int idx = blockIdx.x * blockDim.x + threadIdx.x;
  if (idx >= rows) return;
  const float* hr = h + (size_t)idx * 32;
  float s = 0.0f, d = 0.0f;
#pragma unroll
  for (int t = 0; t < 32; ++t) { s += hr[t] * a[t]; d += hr[t] * a[32 + t]; }
  esrc[idx] = s;
  edst[idx] = d;
}

// Masked-softmax aggregation + ELU; online softmax over j (self-loops
// guarantee a nonempty row; skipping masked j == exp(-1e9 - m) == 0 in f32).
__global__ void gat_attn_kernel(const float* __restrict__ h,
                                const float* __restrict__ esrc,
                                const float* __restrict__ edst,
                                const float* __restrict__ Adj,
                                float* __restrict__ out, int B, int N, int head) {
  int idx = blockIdx.x * blockDim.x + threadIdx.x;
  if (idx >= B * N) return;
  int b = idx / N, i = idx % N;
  float ei = esrc[idx];
  const float* Ar = Adj + ((size_t)b * N + i) * N;
  const float* ed = edst + (size_t)b * N;
  float m = -3.4e38f, s = 0.0f, o[32];
#pragma unroll
  for (int t = 0; t < 32; ++t) o[t] = 0.0f;
  for (int j = 0; j < N; ++j) {
    if (!(Ar[j] > 0.0f)) continue;
    float e = ei + ed[j];
    e = (e > 0.0f) ? e : 0.2f * e;  // leaky_relu(alpha=0.2)
    if (e > m) {
      float f = __expf(m - e);
      s *= f;
#pragma unroll
      for (int t = 0; t < 32; ++t) o[t] *= f;
      m = e;
    }
    float w = __expf(e - m);
    s += w;
    const float* hr = h + ((size_t)b * N + j) * 32;
#pragma unroll
    for (int t = 0; t < 32; ++t) o[t] += w * hr[t];
  }
  float inv = 1.0f / s;
  float* op = out + ((size_t)b * N + i) * 128 + head * 32;
#pragma unroll
  for (int t = 0; t < 32; ++t) {
    float v = o[t] * inv;
    op[t] = (v > 0.0f) ? v : (__expf(v) - 1.0f);  // ELU
  }
}

// Training-mode BatchNorm over batch axis + ReLU; one wave per feature.
__global__ void bn_relu_kernel(float* __restrict__ x, const float* __restrict__ g,
                               const float* __restrict__ b, int M, int N) {
  int feat = (int)((blockIdx.x * blockDim.x + threadIdx.x) >> 5);
  int lane = threadIdx.x & 31;
  if (feat >= N) return;
  float s = 0.0f, s2 = 0.0f;
  for (int r = lane; r < M; r += 32) {
    float v = x[(size_t)r * N + feat];
    s += v; s2 += v * v;
  }
  s = wave_sum(s);
  s2 = wave_sum(s2);
  float mean = s / (float)M;
  float var = s2 / (float)M - mean * mean;
  float rstd = rsqrtf(var + 1e-5f);
  float gg = g[feat], bb = b[feat];
  for (int r = lane; r < M; r += 32) {
    float v = (x[(size_t)r * N + feat] - mean) * rstd * gg + bb;
    x[(size_t)r * N + feat] = fmaxf(v, 0.0f);
  }
}

__global__ void concat_kernel(const float* s0, const float* s1, const float* s2,
                              const float* s3, const float* s4, const float* s5,
                              const float* s6, float* __restrict__ out, int B) {
  int idx = blockIdx.x * blockDim.x + threadIdx.x;
  if (idx >= B * 896) return;
  int b = idx / 896, j = idx % 896;
  const float* srcs[7] = {s0, s1, s2, s3, s4, s5, s6};
  out[idx] = srcs[j >> 7][b * 128 + (j & 127)];
}

__global__ void fc_sigmoid_kernel(const float* __restrict__ h, const float* __restrict__ Wf,
                                  const float* __restrict__ bf, float* __restrict__ p, int B) {
  int b = blockIdx.x * blockDim.x + threadIdx.x;
  if (b >= B) return;
  float s = bf[0];
#pragma unroll 4
  for (int t = 0; t < 128; ++t) s += h[(size_t)b * 128 + t] * Wf[t];
  p[b] = 1.0f / (1.0f + __expf(-s));
}

__global__ void loss_kernel(const float* __restrict__ p, const float* __restrict__ label,
                            float* __restrict__ out, int B) {
  __shared__ float red[512];
  int t = threadIdx.x;
  float pv = p[t];
  float pc = fminf(fmaxf(pv, 1e-7f), 1.0f - 1e-7f);
  float l = label[t];
  red[t] = -(l * __logf(pc) + (1.0f - l) * log1pf(-pc));
  out[t] = pv;
  __syncthreads();
  for (int off = 256; off > 0; off >>= 1) {
    if (t < off) red[t] += red[t + off];
    __syncthreads();
  }
  if (t == 0) out[B] = red[0] / (float)B;
}

// ------------------------------ host driver --------------------------------

static inline void launch_pack_a(const float* A, _Float16* Ap, int M, int K,
                                 hipStream_t s) {
  int n = ((K + 31) >> 5) * (M >> 4) * 32;
  pack_a_kernel<<<(n + 255) / 256, 256, 0, s>>>(A, Ap, M, K);
}

static inline void launch_gemm_packed(const _Float16* Ap, const float* B,
                                      _Float16* Bp, const float* bias, float* C,
                                      int M, int N, int K, int act, hipStream_t s) {
  int Kt = (K + 31) >> 5;
  int nb = Kt * (N >> 4) * 32;
  pack_b_kernel<<<(nb + 255) / 256, 256, 0, s>>>(B, Bp, K, N);
  int slabs = (M >> 4) * (N >> 5);  // one wave per 16x32 slab
  int grid = (slabs + 7) / 8;       // 8 waves per 256-thread block
  if (act)
    gemm_wmma_kernel<1><<<grid, 256, 0, s>>>(Ap, Bp, bias, C, M, N, Kt);
  else
    gemm_wmma_kernel<0><<<grid, 256, 0, s>>>(Ap, Bp, bias, C, M, N, Kt);
}

static inline void launch_gemm(const float* A, const float* B, const float* bias,
                               float* C, int M, int N, int K, int act,
                               _Float16* Ap, _Float16* Bp, hipStream_t s) {
  launch_pack_a(A, Ap, M, K, s);
  launch_gemm_packed(Ap, B, Bp, bias, C, M, N, K, act, s);
}

extern "C" void kernel_launch(void* const* d_in, const int* in_sizes, int n_in,
                              void* d_out, int out_size, void* d_ws, size_t ws_size,
                              hipStream_t stream) {
  (void)in_sizes; (void)n_in; (void)out_size; (void)ws_size;
  const int B = 512, L = 79, E = 128, Ngr = 100;
  const long RT = (long)B * L;   // 40448 token rows (multiple of 16)
  const long GR = (long)B * Ngr; // 51200 node rows  (multiple of 16)

  // ---- input pointer map (setup_inputs() insertion order; params leaves in
  // nested insertion order: emb, enc[6]{Wqkv,bqkv,Wo,bo,W1,b1,W2,b2,
  // ln1_g,ln1_b,ln2_g,ln2_b}, gat[5][4]{W,a}, gproj[2]{W,b,g,bb},
  // fp[5][3]{W,b,g,bb}, proj{W,b,g,bb}, fcW, fcb) ----
  const int*   tok   = (const int*)d_in[0];
  const float* fIn[5]= {(const float*)d_in[1], (const float*)d_in[2],
                        (const float*)d_in[3], (const float*)d_in[4],
                        (const float*)d_in[5]};
  const int    fdim[5] = {1024, 881, 1024, 780, 4860};
  const float* Xg    = (const float*)d_in[6];
  const float* Ag    = (const float*)d_in[7];
  const float* label = (const float*)d_in[8];
  const float* emb   = (const float*)d_in[9];
  const int ENC0   = 10;
  const int GAT0   = ENC0 + 6 * 12;     // 82
  const int GPROJ0 = GAT0 + 5 * 4 * 2;  // 122
  const int FP0    = GPROJ0 + 2 * 4;    // 130
  const int PROJ0  = FP0 + 5 * 3 * 4;   // 190
  const int FCW    = PROJ0 + 4;         // 194
  const int FCB    = FCW + 1;           // 195
  auto P = [&](int i) { return (const float*)d_in[i]; };

  // ---- workspace layout (floats) ----
  float* ws = (float*)d_ws;
  size_t off = 0;
  auto alloc = [&](size_t n) { float* p = ws + off; off += n; return p; };
  float* smiles = alloc((size_t)B * E);
  float* graph  = alloc((size_t)B * E);
  float* fpout[5];
  for (int t = 0; t < 5; ++t) fpout[t] = alloc((size_t)B * E);
  float* comb   = alloc((size_t)B * 896);
  float* hfuse  = alloc((size_t)B * E);
  float* pbuf   = alloc((size_t)B);
  // packed-operand scratch (f16): A up to 40448x256, B up to 12800x1024
  _Float16* apack = (_Float16*)alloc((size_t)RT * 256 / 2 + 64);
  _Float16* bpack = (_Float16*)alloc((size_t)12800 * 1024 / 2 + 64);
  float* big = ws + off;  // large phase-reused region

  // Phase 1 layout (transformer): 40448*(128+384+128+128+256) floats
  float* xbuf  = big;
  float* qkvb  = xbuf + RT * 128;
  float* attnb = qkvb + RT * 384;
  float* ybuf  = attnb + RT * 128;
  float* ffb   = ybuf + RT * 128;

  // ---- SMILES transformer ----
  {
    int total = (int)(RT * E);
    embed_pe_kernel<<<(total + 255) / 256, 256, 0, stream>>>(tok, emb, xbuf, B, L, E);
  }
  for (int layer = 0; layer < 6; ++layer) {
    const int Lp = ENC0 + 12 * layer;
    const float *Wqkv = P(Lp+0), *bqkv = P(Lp+1), *Wo = P(Lp+2), *bo = P(Lp+3);
    const float *W1 = P(Lp+4), *b1 = P(Lp+5), *W2 = P(Lp+6), *b2 = P(Lp+7);
    const float *g1 = P(Lp+8), *bb1 = P(Lp+9), *g2 = P(Lp+10), *bb2 = P(Lp+11);
    launch_gemm(xbuf, Wqkv, bqkv, qkvb, (int)RT, 384, 128, 0, apack, bpack, stream);
    {
      int total = B * 8 * L;
      attention_kernel<<<(total + 255) / 256, 256, 0, stream>>>(qkvb, attnb, B, L);
    }
    launch_gemm(attnb, Wo, bo, ybuf, (int)RT, 128, 128, 0, apack, bpack, stream);
    add_ln_kernel<<<(int)((RT * 32 + 255) / 256), 256, 0, stream>>>(xbuf, ybuf, g1, bb1, (int)RT);
    launch_gemm(xbuf, W1, b1, ffb, (int)RT, 256, 128, 1, apack, bpack, stream);
    launch_gemm(ffb, W2, b2, ybuf, (int)RT, 128, 256, 0, apack, bpack, stream);
    add_ln_kernel<<<(int)((RT * 32 + 255) / 256), 256, 0, stream>>>(xbuf, ybuf, g2, bb2, (int)RT);
  }
  mean_seq_kernel<<<(B * E + 255) / 256, 256, 0, stream>>>(xbuf, smiles, B, L, E);

  // ---- GAT (phase 2 reuses `big`) ----
  float* gA   = big;
  float* gB   = gA + GR * 128;
  float* hb   = gB + GR * 128;   // 4 heads x (GR,32)
  float* esrc = hb + GR * 128;
  float* edst = esrc + 4 * GR;
  float* gp1  = edst + 4 * GR;   // (512,1024) gproj intermediate

  const float* gsrc = Xg;
  float* gdst = gA;
  int din = 65;
  for (int layer = 0; layer < 5; ++layer) {
    launch_pack_a(gsrc, apack, (int)GR, din, stream);  // shared across 4 heads
    for (int hd = 0; hd < 4; ++hd) {
      const float* Wh = P(GAT0 + (layer * 4 + hd) * 2 + 0);
      const float* ah = P(GAT0 + (layer * 4 + hd) * 2 + 1);
      float* hh = hb + (size_t)hd * GR * 32;
      launch_gemm_packed(apack, Wh, bpack, nullptr, hh, (int)GR, 32, din, 0, stream);
      gat_edge_kernel<<<(int)((GR + 255) / 256), 256, 0, stream>>>(
          hh, ah, esrc + hd * GR, edst + hd * GR, (int)GR);
    }
    for (int hd = 0; hd < 4; ++hd) {
      float* hh = hb + (size_t)hd * GR * 32;
      gat_attn_kernel<<<(int)((GR + 255) / 256), 256, 0, stream>>>(
          hh, esrc + hd * GR, edst + hd * GR, Ag, gdst, B, Ngr, hd);
    }
    gsrc = gdst;
    gdst = (gdst == gA) ? gB : gA;
    din = 128;
  }
  const float* gflat = gsrc;  // (B, 12800), ends in gA after 5 layers

  // gproj tower: 12800 -> 1024 -> 128 with BN+ReLU
  launch_gemm(gflat, P(GPROJ0+0), P(GPROJ0+1), gp1, B, 1024, 12800, 0, apack, bpack, stream);
  bn_relu_kernel<<<(1024 * 32 + 255) / 256, 256, 0, stream>>>(gp1, P(GPROJ0+2), P(GPROJ0+3), B, 1024);
  launch_gemm(gp1, P(GPROJ0+4), P(GPROJ0+5), graph, B, 128, 1024, 0, apack, bpack, stream);
  bn_relu_kernel<<<(128 * 32 + 255) / 256, 256, 0, stream>>>(graph, P(GPROJ0+6), P(GPROJ0+7), B, 128);

  // ---- fingerprint towers (phase 3 reuses `big`) ----
  float* t0 = big;                       // (512,512)
  float* t1 = big + (size_t)B * 512;     // (512,256)
  for (int tw = 0; tw < 5; ++tw) {
    const int base = FP0 + tw * 12;
    launch_gemm(fIn[tw], P(base+0), P(base+1), t0, B, 512, fdim[tw], 0, apack, bpack, stream);
    bn_relu_kernel<<<(512 * 32 + 255) / 256, 256, 0, stream>>>(t0, P(base+2), P(base+3), B, 512);
    launch_gemm(t0, P(base+4), P(base+5), t1, B, 256, 512, 0, apack, bpack, stream);
    bn_relu_kernel<<<(256 * 32 + 255) / 256, 256, 0, stream>>>(t1, P(base+6), P(base+7), B, 256);
    launch_gemm(t1, P(base+8), P(base+9), fpout[tw], B, 128, 256, 0, apack, bpack, stream);
    bn_relu_kernel<<<(128 * 32 + 255) / 256, 256, 0, stream>>>(fpout[tw], P(base+10), P(base+11), B, 128);
  }

  // ---- fusion head ----
  concat_kernel<<<(B * 896 + 255) / 256, 256, 0, stream>>>(
      smiles, graph, fpout[0], fpout[1], fpout[2], fpout[3], fpout[4], comb, B);
  launch_gemm(comb, P(PROJ0+0), P(PROJ0+1), hfuse, B, 128, 896, 0, apack, bpack, stream);
  bn_relu_kernel<<<(128 * 32 + 255) / 256, 256, 0, stream>>>(hfuse, P(PROJ0+2), P(PROJ0+3), B, 128);
  fc_sigmoid_kernel<<<(B + 255) / 256, 256, 0, stream>>>(hfuse, P(FCW), P(FCB), pbuf, B);
  loss_kernel<<<1, 512, 0, stream>>>(pbuf, label, (float*)d_out, B);
}
